// Model_39977555591259
// MI455X (gfx1250) — matrix operand order repaired and checked by reading, not compiled, not run on hardware
//
#include <hip/hip_runtime.h>

typedef _Float16 v16h __attribute__((ext_vector_type(16)));
typedef _Float16 v8h  __attribute__((ext_vector_type(8)));
typedef float    v8f  __attribute__((ext_vector_type(8)));

#define C_DIM   256
#define C2_DIM  512
#define H2_DIM  80
#define W2_DIM  80
#define OUT_DIM 255
#define BATCH   4
#define HW2     (H2_DIM * W2_DIM)      /* 6400  */
#define HW1     (160 * 160)            /* 25600 */
#define P2_TOT  (BATCH * HW2)          /* 25600 */

/* ---- workspace layout (bytes) ---- */
#define OFF_W1ALL 0u                          /* [512 out][512 in] f16: rows 0..255 = Wq', 256..511 = M2 (row 511 zero) */
#define SZ_W1ALL  (512u * 512u * 2u)
#define OFF_W2ALL (OFF_W1ALL + SZ_W1ALL)      /* [512 out][256 in] f16: rows 0..255 = Wk', 256..511 = Mv (row 511 zero) */
#define SZ_W2ALL  (512u * 256u * 2u)
#define OFF_WVP   (OFF_W2ALL + SZ_W2ALL)      /* Wv' [256][256] f32 scratch */
#define SZ_WVP    (256u * 256u * 4u)
#define OFF_BQ    (OFF_WVP + SZ_WVP)          /* bq' [256] f32 */
#define OFF_BK    (OFF_BQ + 1024u)            /* bk' [256] f32 */
#define OFF_BF    (OFF_BK + 1024u)            /* bF  [256] f32 */
#define OFF_BVP   (OFF_BF + 1024u)            /* bv' [256] f32 scratch */
#define OFF_QS    (OFF_BVP + 1024u)           /* QS [25600][512] f32 : [.,0:256)=q, [.,256:512)=s2 */

/* =================== K0a: Wv' = conv1_w^T compose wv ; bv' =================== */
__global__ void clla_k0a(const float* __restrict__ conv1_w, const float* __restrict__ conv1_b,
                         const float* __restrict__ wv, const float* __restrict__ bv,
                         char* __restrict__ ws) {
    int gid = blockIdx.x * 256 + threadIdx.x;
    float* Wvp = (float*)(ws + OFF_WVP);
    float* bvp = (float*)(ws + OFF_BVP);
    if (gid < 256 * 256) {
        int o = gid >> 8, i = gid & 255;
        float s = 0.f;
        for (int c = 0; c < 256; ++c) s += conv1_w[c * 256 + i] * wv[c * 256 + o];
        Wvp[o * 256 + i] = s;   /* Wv'[o][i] */
    } else if (gid < 256 * 256 + 256) {
        int o = gid - 256 * 256;
        float s = bv[o];
        for (int c = 0; c < 256; ++c) s += conv1_b[c] * wv[c * 256 + o];
        bvp[o] = s;
    }
}

/* =================== K0b: all fused weights / biases =================== */
__global__ void clla_k0b(const float* __restrict__ conv1_w, const float* __restrict__ conv1_b,
                         const float* __restrict__ conv2_w, const float* __restrict__ conv2_b,
                         const float* __restrict__ wq, const float* __restrict__ bq,
                         const float* __restrict__ wk, const float* __restrict__ bk,
                         const float* __restrict__ det_w, const float* __restrict__ det_b,
                         char* __restrict__ ws) {
    const int TOT1 = 512 * 512;
    const int TOT2 = TOT1 + 512 * 256;
    int gid = blockIdx.x * 256 + threadIdx.x;
    _Float16* W1 = (_Float16*)(ws + OFF_W1ALL);
    _Float16* W2 = (_Float16*)(ws + OFF_W2ALL);
    const float* Wvp = (const float*)(ws + OFF_WVP);
    const float* bvp = (const float*)(ws + OFF_BVP);
    float* bqf = (float*)(ws + OFF_BQ);
    float* bkf = (float*)(ws + OFF_BK);
    float* bFf = (float*)(ws + OFF_BF);

    if (gid < TOT1) {
        int n = gid >> 9, i2 = gid & 511;
        float s = 0.f;
        if (n < 256) {                       /* Wq' */
            for (int c = 0; c < 256; ++c) s += conv2_w[c * 512 + i2] * wq[c * 256 + n];
        } else if (n < 256 + OUT_DIM) {      /* M2 = 0.5*det_w.conv2_w */
            int o = n - 256;
            for (int c = 0; c < 256; ++c) s += det_w[o * 256 + c] * conv2_w[c * 512 + i2];
            s *= 0.5f;
        }
        W1[(size_t)n * 512 + i2] = (_Float16)s;
    } else if (gid < TOT2) {
        int g = gid - TOT1;
        int n = g >> 8, i = g & 255;
        float s = 0.f;
        if (n < 256) {                       /* Wk' */
            for (int c = 0; c < 256; ++c) s += conv1_w[c * 256 + i] * wk[c * 256 + n];
        } else if (n < 256 + OUT_DIM) {      /* Mv = 0.5*det_w.Wv' */
            int o = n - 256;
            for (int c = 0; c < 256; ++c) s += det_w[o * 256 + c] * Wvp[c * 256 + i];
            s *= 0.5f;
        }
        W2[(size_t)n * 256 + i] = (_Float16)s;
    } else if (gid < TOT2 + 256) {
        int o = gid - TOT2;
        float s = bq[o];
        for (int c = 0; c < 256; ++c) s += conv2_b[c] * wq[c * 256 + o];
        bqf[o] = s;
    } else if (gid < TOT2 + 512) {
        int o = gid - TOT2 - 256;
        float s = bk[o];
        for (int c = 0; c < 256; ++c) s += conv1_b[c] * wk[c * 256 + o];
        bkf[o] = s;
    } else if (gid < TOT2 + 768) {
        int o = gid - TOT2 - 512;
        float s = 0.f;
        if (o < OUT_DIM) {
            s = det_b[o];
            float a = 0.f;
            for (int c = 0; c < 256; ++c) a += det_w[o * 256 + c] * (bvp[c] + conv2_b[c]);
            s += 0.5f * a;
        }
        bFf[o] = s;
    }
}

/* ---- GEMM tile constants ---- */
#define SA 40                                 /* halves per LDS row (80 B) */
#define A_BYTES (64 * SA * 2)                 /* 5120  */
#define B_OFFB  A_BYTES                       /* Blds at +5120, [512][SA] */
#define STG_BYTES (A_BYTES + 512 * SA * 2)    /* 46080 per buffer */

/* async 16B global -> LDS copy (ASYNCcnt-tracked, no VGPR staging) */
__device__ __forceinline__ void async_g2l_b128(unsigned int lds_off, const void* gaddr) {
    asm volatile("global_load_async_to_lds_b128 %0, %1, off"
                 :: "v"(lds_off), "v"(gaddr) : "memory");
}
__device__ __forceinline__ void wait_async0() {
    asm volatile("s_wait_asynccnt 0" ::: "memory");
}

/* =================== K1: x2 -> QS  (M=25600, N=512, K=512) =================== */
__global__ __launch_bounds__(256) void clla_x2_gemm(const float* __restrict__ x2,
                                                    char* __restrict__ ws) {
    extern __shared__ char smem[];            /* 2 x STG_BYTES (double buffer) */
    const _Float16* W1 = (const _Float16*)(ws + OFF_W1ALL);
    const float* bq = (const float*)(ws + OFF_BQ);
    float* QS = (float*)(ws + OFF_QS);

    const int t = threadIdx.x;
    const int m0 = blockIdx.x * 64;
    const int b = m0 / HW2;
    const int hw0 = m0 % HW2;
    const int wid = t >> 5, lane = t & 31;
    const int nbase = wid * 64;
    const int g = lane >> 4;

    float areg[8];
    auto loadA = [&](int ks) {
#pragma unroll
        for (int i = 0; i < 8; ++i) {
            int idx = i * 256 + t;
            int k = idx >> 6, m = idx & 63;
            areg[i] = x2[((size_t)b * C2_DIM + (ks * 32 + k)) * HW2 + hw0 + m];
        }
    };
    auto storeA = [&](int buf) {
        _Float16* Al = (_Float16*)(smem + buf * STG_BYTES);
#pragma unroll
        for (int i = 0; i < 8; ++i) {
            int idx = i * 256 + t;
            int k = idx >> 6, m = idx & 63;
            Al[m * SA + k] = (_Float16)areg[i];
        }
    };
    auto asyncB = [&](int ks, int buf) {
        char* Bb = smem + buf * STG_BYTES + B_OFFB;
#pragma unroll
        for (int r = 0; r < 2; ++r) {
            int n = 2 * t + r;
            const char* gp = (const char*)(W1 + (size_t)n * 512 + ks * 32);
            unsigned int lo = (unsigned int)(unsigned long long)(Bb + n * (SA * 2));
#pragma unroll
            for (int c = 0; c < 4; ++c)
                async_g2l_b128(lo + 16u * c, gp + 16 * c);
        }
    };

    v8f acc[4][4];
#pragma unroll
    for (int mt = 0; mt < 4; ++mt)
#pragma unroll
        for (int nt = 0; nt < 4; ++nt) { v8f z = {}; acc[mt][nt] = z; }

    /* prologue: tile 0 -> buffer 0 */
    loadA(0); storeA(0); asyncB(0, 0);
    wait_async0();
    __syncthreads();

    for (int ks = 0; ks < 16; ++ks) {
        const int cur = ks & 1;
        const bool pre = (ks + 1 < 16);
        if (pre) { loadA(ks + 1); asyncB(ks + 1, cur ^ 1); }

        const _Float16* Al = (const _Float16*)(smem + cur * STG_BYTES);
        const _Float16* Bl = (const _Float16*)(smem + cur * STG_BYTES + B_OFFB);
        union { v16h v; v8h h[2]; } af, bf[4];
#pragma unroll
        for (int nt = 0; nt < 4; ++nt) {      /* B frag: K = 16g + e */
            const _Float16* p = Bl + (nbase + nt * 16 + (lane & 15)) * SA + 16 * g;
            bf[nt].h[0] = *(const v8h*)p;
            bf[nt].h[1] = *(const v8h*)(p + 8);
        }
#pragma unroll
        for (int mt = 0; mt < 4; ++mt) {      /* A frag: K = 16g + e */
            const _Float16* p = Al + (mt * 16 + (lane & 15)) * SA;
            af.h[0] = *(const v8h*)(p + 16 * g);
            af.h[1] = *(const v8h*)(p + 16 * g + 8);
#pragma unroll
            for (int nt = 0; nt < 4; ++nt)
                acc[mt][nt] = __builtin_amdgcn_wmma_f32_16x16x32_f16(
                    false, af.v, false, bf[nt].v, (short)0, acc[mt][nt], false, false);
        }

        if (pre) storeA(cur ^ 1);
        wait_async0();
        __syncthreads();
    }

#pragma unroll
    for (int nt = 0; nt < 4; ++nt) {
        int n = nbase + nt * 16 + (lane & 15);
        float bias = (n < 256) ? bq[n] : 0.f;
#pragma unroll
        for (int mt = 0; mt < 4; ++mt) {
            int mb = mt * 16 + g * 8;
#pragma unroll
            for (int r = 0; r < 8; ++r)
                QS[(size_t)(m0 + mb + r) * 512 + n] = acc[mt][nt][r] + bias;
        }
    }
}

/* =================== K2: x1 -> K/Yv GEMM + attention + output =================== */
#define KVS 521                               /* f32 row stride, conflict-free */
__global__ __launch_bounds__(256) void clla_x1_attn(const float* __restrict__ x1,
                                                    float* __restrict__ out,
                                                    char* __restrict__ ws) {
    extern __shared__ char smem[];
    float* KV = (float*)smem;                              /* [64][KVS] (overlays staging) */
    float* dotb = (float*)(smem + 64 * KVS * 4);           /* [64][4]  */
    float* attl = dotb + 256;                              /* [16][4]  */

    const _Float16* W2 = (const _Float16*)(ws + OFF_W2ALL);
    const float* bk = (const float*)(ws + OFF_BK);
    const float* bF = (const float*)(ws + OFF_BF);
    const float* QS = (const float*)(ws + OFF_QS);

    const int t = threadIdx.x;
    const int tile = blockIdx.x;                 /* 1600 tiles of 16 x2-pixels */
    const int b = tile / 400;
    const int rem = tile % 400;
    const int h2 = rem / 5;
    const int w20 = (rem % 5) * 16;
    const int p20 = b * HW2 + h2 * W2_DIM + w20;

    const int wid = t >> 5, lane = t & 31;
    const int nbase = wid * 64;
    const int g = lane >> 4;

    float areg[8];
    auto loadA = [&](int ks) {
#pragma unroll
        for (int i = 0; i < 8; ++i) {
            int idx = i * 256 + t;
            int k = idx >> 6, m = idx & 63;
            int p2l = m >> 2, j = m & 3;
            int h1 = 2 * h2 + (j >> 1);
            int w1 = 2 * (w20 + p2l) + (j & 1);
            areg[i] = x1[((size_t)b * C_DIM + (ks * 32 + k)) * HW1 + h1 * 160 + w1];
        }
    };
    auto storeA = [&](int buf) {
        _Float16* Al = (_Float16*)(smem + buf * STG_BYTES);
#pragma unroll
        for (int i = 0; i < 8; ++i) {
            int idx = i * 256 + t;
            int k = idx >> 6, m = idx & 63;
            Al[m * SA + k] = (_Float16)areg[i];
        }
    };
    auto asyncB = [&](int ks, int buf) {
        char* Bb = smem + buf * STG_BYTES + B_OFFB;
#pragma unroll
        for (int r = 0; r < 2; ++r) {
            int n = 2 * t + r;
            const char* gp = (const char*)(W2 + (size_t)n * 256 + ks * 32);
            unsigned int lo = (unsigned int)(unsigned long long)(Bb + n * (SA * 2));
#pragma unroll
            for (int c = 0; c < 4; ++c)
                async_g2l_b128(lo + 16u * c, gp + 16 * c);
        }
    };

    v8f acc[4][4];
#pragma unroll
    for (int mt = 0; mt < 4; ++mt)
#pragma unroll
        for (int nt = 0; nt < 4; ++nt) { v8f z = {}; acc[mt][nt] = z; }

    loadA(0); storeA(0); asyncB(0, 0);
    wait_async0();
    __syncthreads();

    for (int ks = 0; ks < 8; ++ks) {
        const int cur = ks & 1;
        const bool pre = (ks + 1 < 8);
        if (pre) { loadA(ks + 1); asyncB(ks + 1, cur ^ 1); }

        const _Float16* Al = (const _Float16*)(smem + cur * STG_BYTES);
        const _Float16* Bl = (const _Float16*)(smem + cur * STG_BYTES + B_OFFB);
        union { v16h v; v8h h[2]; } af, bf[4];
#pragma unroll
        for (int nt = 0; nt < 4; ++nt) {
            const _Float16* p = Bl + (nbase + nt * 16 + (lane & 15)) * SA + 16 * g;
            bf[nt].h[0] = *(const v8h*)p;
            bf[nt].h[1] = *(const v8h*)(p + 8);
        }
#pragma unroll
        for (int mt = 0; mt < 4; ++mt) {
            const _Float16* p = Al + (mt * 16 + (lane & 15)) * SA;
            af.h[0] = *(const v8h*)(p + 16 * g);
            af.h[1] = *(const v8h*)(p + 16 * g + 8);
#pragma unroll
            for (int nt = 0; nt < 4; ++nt)
                acc[mt][nt] = __builtin_amdgcn_wmma_f32_16x16x32_f16(
                    false, af.v, false, bf[nt].v, (short)0, acc[mt][nt], false, false);
        }

        if (pre) storeA(cur ^ 1);
        wait_async0();
        __syncthreads();
    }

    /* spill K (cols 0:256, +bk') and Yv (cols 256:512) to LDS */
#pragma unroll
    for (int nt = 0; nt < 4; ++nt) {
        int n = nbase + nt * 16 + (lane & 15);
        float bias = (n < 256) ? bk[n] : 0.f;
#pragma unroll
        for (int mt = 0; mt < 4; ++mt) {
            int mb = mt * 16 + g * 8;
#pragma unroll
            for (int r = 0; r < 8; ++r)
                KV[(mb + r) * KVS + n] = acc[mt][nt][r] + bias;
        }
    }
    __syncthreads();

    /* dots: (q . k)/2, 4 channel-interleaved partials per window row */
    {
        int row = t >> 2, ch = t & 3;
        int p2l = row >> 2;
        const float* q = QS + (size_t)(p20 + p2l) * 512;
        float s = 0.f;
        for (int cc = 0; cc < 64; ++cc) {
            int c = ch + 4 * cc;
            s += q[c] * KV[row * KVS + c];
        }
        dotb[row * 4 + ch] = 0.5f * s;
    }
    __syncthreads();

    /* softmax of irr = 2*mean - dots, per x2 pixel */
    if (t < 16) {
        float d[4];
#pragma unroll
        for (int j = 0; j < 4; ++j) {
            const float* p = dotb + (t * 4 + j) * 4;
            d[j] = p[0] + p[1] + p[2] + p[3];
        }
        float mean = 0.25f * (d[0] + d[1] + d[2] + d[3]);
        float irr[4], mx = -1e30f;
#pragma unroll
        for (int j = 0; j < 4; ++j) { irr[j] = 2.f * mean - d[j]; mx = fmaxf(mx, irr[j]); }
        float e[4], s = 0.f;
#pragma unroll
        for (int j = 0; j < 4; ++j) { e[j] = __expf(irr[j] - mx); s += e[j]; }
        float inv = 1.f / s;
#pragma unroll
        for (int j = 0; j < 4; ++j) attl[t * 4 + j] = e[j] * inv;
    }
    __syncthreads();

    /* output: bF + s2 + sum_j att_j * Yv_j  -> [B,255,80,80] */
    for (int i = 0; i < 16; ++i) {
        int idx = i * 256 + t;
        if (idx < 16 * OUT_DIM) {
            int o = idx >> 4, p2l = idx & 15;
            int rb = p2l * 4;
            float a0 = attl[rb + 0], a1 = attl[rb + 1], a2 = attl[rb + 2], a3 = attl[rb + 3];
            float val = bF[o] + QS[(size_t)(p20 + p2l) * 512 + 256 + o]
                      + a0 * KV[(rb + 0) * KVS + 256 + o]
                      + a1 * KV[(rb + 1) * KVS + 256 + o]
                      + a2 * KV[(rb + 2) * KVS + 256 + o]
                      + a3 * KV[(rb + 3) * KVS + 256 + o];
            out[(((size_t)b * OUT_DIM + o) * H2_DIM + h2) * W2_DIM + w20 + p2l] = val;
        }
    }
}

extern "C" void kernel_launch(void* const* d_in, const int* in_sizes, int n_in,
                              void* d_out, int out_size, void* d_ws, size_t ws_size,
                              hipStream_t stream) {
    const float* x1      = (const float*)d_in[0];
    const float* x2      = (const float*)d_in[1];
    const float* conv1_w = (const float*)d_in[2];
    const float* conv1_b = (const float*)d_in[3];
    const float* conv2_w = (const float*)d_in[4];
    const float* conv2_b = (const float*)d_in[5];
    const float* wq      = (const float*)d_in[6];
    const float* bq      = (const float*)d_in[7];
    const float* wk      = (const float*)d_in[8];
    const float* bk      = (const float*)d_in[9];
    const float* wv      = (const float*)d_in[10];
    const float* bv      = (const float*)d_in[11];
    const float* det_w   = (const float*)d_in[12];
    const float* det_b   = (const float*)d_in[13];
    float* out = (float*)d_out;
    char* ws = (char*)d_ws;

    /* weight fusion */
    clla_k0a<<<257, 256, 0, stream>>>(conv1_w, conv1_b, wv, bv, ws);
    clla_k0b<<<1539, 256, 0, stream>>>(conv1_w, conv1_b, conv2_w, conv2_b,
                                       wq, bq, wk, bk, det_w, det_b, ws);
    /* x2 path: Q + S2 (double-buffered, async B tiles) */
    clla_x2_gemm<<<P2_TOT / 64, 256, 2 * STG_BYTES, stream>>>(x2, ws);
    /* x1 path GEMM fused with attention + det head */
    clla_x1_attn<<<1600, 256, 64 * KVS * 4 + 1024 + 256, stream>>>(x1, out, ws);
}
